// Decoder_2834678415878
// MI455X (gfx1250) — matrix operand order repaired and checked by reading, not compiled, not run on hardware
//
#include <hip/hip_runtime.h>

// ---------------------------------------------------------------------------
// Types for CDNA5 WMMA (wave32, 16x16x32 bf16 -> f32)
// ---------------------------------------------------------------------------
typedef __attribute__((ext_vector_type(16))) __bf16 v16bf;
typedef __attribute__((ext_vector_type(8)))  float  v8f;

#define BM 128
#define BN 128
#define BK 32
#define LDA 40   // As row stride in bf16 elements (padded, even)
#define LDB 36   // Bs row stride in bf16 elements (padded, even)

// ---------------------------------------------------------------------------
// Generic GEMM: C[M,N] = A[M,K] (f32, row-major) * W[K,N] (f32, row-major) + bias
// A/W converted to bf16 in LDS; f32 accumulation via v_wmma_f32_16x16x32_bf16.
// Software-pipelined: global loads for tile t+1 overlap WMMA compute of tile t.
// Requirements: M % 128 == 0, K % 32 == 0, N % 4 == 0 (all true here); N guarded.
// Block: 256 threads = 8 waves; block tile 128(M) x 128(N); wave tile 32 x 64.
// ---------------------------------------------------------------------------
__global__ __launch_bounds__(256)
void gemm_bf16_wmma(float* __restrict__ C, const float* __restrict__ A,
                    const float* __restrict__ W, const float* __restrict__ bias,
                    int M, int K, int N) {
  __shared__ __align__(16) __bf16 As[BM * LDA];
  __shared__ __align__(16) __bf16 Bs[BN * LDB];

  const int tid  = threadIdx.x;
  const int lane = tid & 31;
  const int wave = tid >> 5;     // 0..7
  const int wm   = wave & 3;     // M-group (32 rows each)
  const int wn   = wave >> 2;    // N-group (64 cols each)
  const int lid  = lane & 15;
  const int half = lane >> 4;

  const int m0 = blockIdx.y * BM;
  const int n0 = blockIdx.x * BN;

  // ---- Per-thread staging addresses (k0-independent parts), hoisted ----
  // A tile (128 x 32 f32): 1024 float4, 4 per thread.
  const float* pa[4];
  __bf16* sa[4];
#pragma unroll
  for (int r = 0; r < 4; ++r) {
    int idx = r * 256 + tid;
    int m  = idx >> 3;           // 8 float4 per 32-wide row
    int kq = (idx & 7) << 2;
    pa[r] = A + (size_t)(m0 + m) * K + kq;
    sa[r] = &As[m * LDA + kq];
  }
  // B tile (32 x 128 f32): unit = 2 consecutive k-rows x 4 cols; 2 per thread.
  const float* pb[2];
  __bf16* sb[2];
  bool bvld[2];
#pragma unroll
  for (int r = 0; r < 2; ++r) {
    int u  = r * 256 + tid;
    int k  = (u >> 5) << 1;      // even k
    int nq = (u & 31) << 2;      // multiple-of-4 column
    bvld[r] = (n0 + nq < N);     // N % 4 == 0 -> whole float4 in/out
    pb[r] = W + (size_t)k * N + (n0 + nq);
    sb[r] = &Bs[nq * LDB + k];
  }

  float4 ra[4];
  float4 rb[2][2];

  auto load_tiles = [&](int k0) {
#pragma unroll
    for (int r = 0; r < 4; ++r)
      ra[r] = *(const float4*)(pa[r] + k0);
#pragma unroll
    for (int r = 0; r < 2; ++r) {
      rb[r][0] = make_float4(0.f, 0.f, 0.f, 0.f);
      rb[r][1] = make_float4(0.f, 0.f, 0.f, 0.f);
      if (bvld[r]) {
        rb[r][0] = *(const float4*)(pb[r] + (size_t)k0 * N);
        rb[r][1] = *(const float4*)(pb[r] + (size_t)(k0 + 1) * N);
      }
    }
  };

  auto stage_tiles = [&]() {
#pragma unroll
    for (int r = 0; r < 4; ++r) {
      union { __bf16 h[4]; unsigned long long u; } q;
      q.h[0] = (__bf16)ra[r].x; q.h[1] = (__bf16)ra[r].y;
      q.h[2] = (__bf16)ra[r].z; q.h[3] = (__bf16)ra[r].w;
      *(unsigned long long*)sa[r] = q.u;
    }
#pragma unroll
    for (int r = 0; r < 2; ++r) {
      const float a0[4] = {rb[r][0].x, rb[r][0].y, rb[r][0].z, rb[r][0].w};
      const float a1[4] = {rb[r][1].x, rb[r][1].y, rb[r][1].z, rb[r][1].w};
#pragma unroll
      for (int j = 0; j < 4; ++j) {
        union { __bf16 h[2]; unsigned u32v; } p;
        p.h[0] = (__bf16)a0[j];  // k
        p.h[1] = (__bf16)a1[j];  // k+1
        *(unsigned*)(sb[r] + j * LDB) = p.u32v;
      }
    }
  };

  v8f acc[2][4] = {};

  load_tiles(0);
  for (int k0 = 0; k0 < K; k0 += BK) {
    stage_tiles();               // waits on in-flight loads, converts, stores
    __syncthreads();

    if (k0 + BK < K)             // issue next tile's loads; no wait here
      load_tiles(k0 + BK);

    // A fragments: lanes 0-15 -> K 0..7 & 16..23 ; lanes 16-31 -> K 8..15 & 24..31
    v16bf afrag[2];
#pragma unroll
    for (int mi = 0; mi < 2; ++mi) {
      int m = wm * 32 + mi * 16 + lid;
#pragma unroll
      for (int v = 0; v < 8; ++v) {
        int k = ((v < 4) ? (2 * v) : (8 + 2 * v)) + 8 * half;
        const __bf16* p = &As[m * LDA + k];
        afrag[mi][2 * v]     = p[0];
        afrag[mi][2 * v + 1] = p[1];
      }
    }
#pragma unroll
    for (int j = 0; j < 4; ++j) {
      int n = wn * 64 + j * 16 + lid;
      v16bf bfrag;  // lanes 0-15: K 0..7 & 16..23 ; lanes 16-31: K 8..15 & 24..31 (N = lid)
#pragma unroll
      for (int v = 0; v < 8; ++v) {
        int k = ((v < 4) ? (2 * v) : (8 + 2 * v)) + 8 * half;
        const __bf16* p = &Bs[n * LDB + k];
        bfrag[2 * v]     = p[0];
        bfrag[2 * v + 1] = p[1];
      }
      acc[0][j] = __builtin_amdgcn_wmma_f32_16x16x32_bf16(
          false, afrag[0], false, bfrag, (short)0, acc[0][j], false, false);
      acc[1][j] = __builtin_amdgcn_wmma_f32_16x16x32_bf16(
          false, afrag[1], false, bfrag, (short)0, acc[1][j], false, false);
    }
    __syncthreads();
  }

  // Write-out: C/D layout — VGPR v holds M = v + 8*half, N = lid
#pragma unroll
  for (int j = 0; j < 4; ++j) {
    int n = n0 + wn * 64 + j * 16 + lid;
    if (n < N) {
      float bv = bias ? bias[n] : 0.0f;
#pragma unroll
      for (int mi = 0; mi < 2; ++mi) {
#pragma unroll
        for (int v = 0; v < 8; ++v) {
          int m = m0 + wm * 32 + mi * 16 + v + 8 * half;
          C[(size_t)m * N + n] = acc[mi][j][v] + bv;
        }
      }
    }
  }
}

// ---------------------------------------------------------------------------
// Pixel-shuffle (2x2) + layernorm + concat with encoder skip.
// y: [B*Hpre*Hpre, 4*D] (PE GEMM output), enc: [B*2H*2W, D]
// cat: [B*2H*2W, 2*D]  (first D = LN(shuffled y), second D = enc)
// One wave per output token. Requires D <= 384.
// ---------------------------------------------------------------------------
__global__ __launch_bounds__(256)
void shuffle_ln_cat(float* __restrict__ cat, const float* __restrict__ y,
                    const float* __restrict__ enc, const float* __restrict__ g,
                    const float* __restrict__ bln, int Hpre, int D) {
  const int wave = threadIdx.x >> 5;
  const int lane = threadIdx.x & 31;
  const int H2 = Hpre * 2;
  const int t = blockIdx.x * 8 + wave;  // token in [0, B*H2*H2)

  const int b   = t / (H2 * H2);
  const int rem = t % (H2 * H2);
  const int h2 = rem / H2, w2 = rem % H2;
  const int h = h2 >> 1, r = h2 & 1, w = w2 >> 1, s = w2 & 1;
  const size_t srcRow = (size_t)(b * Hpre + h) * Hpre + w;
  const float* yp = y + srcRow * (size_t)(4 * D) + (size_t)(2 * r + s) * D;

  float vals[12];
  float sum = 0.0f, sq = 0.0f;
  int cnt = 0;
  for (int c = lane; c < D; c += 32) {
    float v = yp[c];
    vals[cnt++] = v;
    sum += v;
    sq  += v * v;
  }
#pragma unroll
  for (int o = 16; o > 0; o >>= 1) {
    sum += __shfl_xor(sum, o, 32);
    sq  += __shfl_xor(sq,  o, 32);
  }
  const float mu  = sum / (float)D;
  const float var = sq / (float)D - mu * mu;
  const float inv = rsqrtf(var + 1e-5f);

  float* cp = cat + (size_t)t * (2 * D);
  cnt = 0;
  for (int c = lane; c < D; c += 32) {
    cp[c]     = (vals[cnt++] - mu) * inv * g[c] + bln[c];
    cp[D + c] = enc[(size_t)t * D + c];
  }
}

// ---------------------------------------------------------------------------
// Neighborhood attention core. One wave per token.
// S_k = (Q_i . K_{nbr[i][k]} + bias[b][k]) / sqrt(D)
// softmax over the *dense* N-length row whose non-neighbor entries are 0:
//   p_k = exp(S_k - m) / ( sum_k exp(S_k - m) + (N-8) * exp(-m) ),  m = max(0, max_k S_k)
// out_i = sum_k p_k * V_{nbr[i][k]}
// ---------------------------------------------------------------------------
__global__ __launch_bounds__(256)
void na_attend(float* __restrict__ out, const float* __restrict__ Q,
               const float* __restrict__ Kt, const float* __restrict__ Vt,
               const int* __restrict__ nbr, const float* __restrict__ bias,
               int Ntok, int D) {
  const int wave = threadIdx.x >> 5;
  const int lane = threadIdx.x & 31;
  const int t = blockIdx.x * 8 + wave;  // token in [0, B*Ntok)
  const int b = t / Ntok;
  const int i = t % Ntok;
  const float invs = rsqrtf((float)D);

  int nb[8];
#pragma unroll
  for (int k = 0; k < 8; ++k) nb[k] = nbr[i * 8 + k];

  const float* qp = Q + (size_t)t * D;
  float sv[8];
#pragma unroll
  for (int k = 0; k < 8; ++k) {
    const float* kp = Kt + ((size_t)b * Ntok + nb[k]) * (size_t)D;
    float d = 0.0f;
    for (int c = lane; c < D; c += 32) d += qp[c] * kp[c];
#pragma unroll
    for (int o = 16; o > 0; o >>= 1) d += __shfl_xor(d, o, 32);
    sv[k] = (d + bias[b * 8 + k]) * invs;
  }

  float m = 0.0f;  // background zeros participate in the max
#pragma unroll
  for (int k = 0; k < 8; ++k) m = fmaxf(m, sv[k]);
  float denom = (float)(Ntok - 8) * __expf(-m);
  float p[8];
#pragma unroll
  for (int k = 0; k < 8; ++k) {
    p[k] = __expf(sv[k] - m);
    denom += p[k];
  }
  const float rd = 1.0f / denom;

  for (int c = lane; c < D; c += 32) {
    float acc = 0.0f;
#pragma unroll
    for (int k = 0; k < 8; ++k)
      acc += p[k] * Vt[((size_t)b * Ntok + nb[k]) * (size_t)D + c];
    out[(size_t)t * D + c] = acc * rd;
  }
}

// ---------------------------------------------------------------------------
// Host-side orchestration
// ---------------------------------------------------------------------------
extern "C" void kernel_launch(void* const* d_in, const int* in_sizes, int n_in,
                              void* d_out, int out_size, void* d_ws, size_t ws_size,
                              hipStream_t stream) {
  (void)in_sizes; (void)n_in; (void)out_size; (void)ws_size;

  const float* x0     = (const float*)d_in[0];
  const float* enc[3] = {(const float*)d_in[1], (const float*)d_in[2],
                         (const float*)d_in[3]};
  const int*   nbr[3] = {(const int*)d_in[4], (const int*)d_in[5],
                         (const int*)d_in[6]};

  // params flattened jax-style (sorted keys):
  // na0..na2: {bias, bk, bo, bq, bv, wk, wo, wq, wv}
  // pe0..pe2: {ln_b, ln_g, w_exp}
  // skip0..skip2: {b, w}
  struct NA { const float *bias, *bk, *bo, *bq, *bv, *wk, *wo, *wq, *wv; } na[3];
  struct PE { const float *ln_b, *ln_g, *w_exp; } pe[3];
  struct SK { const float *b, *w; } sk[3];
  int idx = 7;
  for (int i = 0; i < 3; ++i) {
    na[i].bias = (const float*)d_in[idx++];
    na[i].bk   = (const float*)d_in[idx++];
    na[i].bo   = (const float*)d_in[idx++];
    na[i].bq   = (const float*)d_in[idx++];
    na[i].bv   = (const float*)d_in[idx++];
    na[i].wk   = (const float*)d_in[idx++];
    na[i].wo   = (const float*)d_in[idx++];
    na[i].wq   = (const float*)d_in[idx++];
    na[i].wv   = (const float*)d_in[idx++];
  }
  for (int i = 0; i < 3; ++i) {
    pe[i].ln_b  = (const float*)d_in[idx++];
    pe[i].ln_g  = (const float*)d_in[idx++];
    pe[i].w_exp = (const float*)d_in[idx++];
  }
  for (int i = 0; i < 3; ++i) {
    sk[i].b = (const float*)d_in[idx++];
    sk[i].w = (const float*)d_in[idx++];
  }

  // Workspace layout: six 12.58MB regions (3,145,728 f32 each)
  const size_t SZ = 12582912;  // bytes
  char* ws = (char*)d_ws;
  float* bufPE  = (float*)(ws);            // PE GEMM out; reused as attn out
  float* bufCat = (float*)(ws + 1 * SZ);   // concat [T, 2D] (spans R1+R2)
  float* bufQ   = (float*)(ws + 1 * SZ);   // reuses cat region (cat dead by then)
  float* bufK   = (float*)(ws + 2 * SZ);
  float* bufV   = (float*)(ws + 3 * SZ);
  float* R4     = (float*)(ws + 4 * SZ);
  float* R5     = (float*)(ws + 5 * SZ);

  const int Bsz = 8;
  const int PE_D[3] = {768, 384, 192};

  const float* xcur = x0;
  for (int i = 0; i < 3; ++i) {
    const int Dp = PE_D[i], D = Dp / 2;
    const int Hpre = 8 << i, H2 = Hpre * 2;
    const int Mpre = Bsz * Hpre * Hpre;  // tokens before expand
    const int T    = Bsz * H2 * H2;      // tokens after expand
    const int Ntok = H2 * H2;

    float* xs   = (i <= 1) ? R4 : R5;                             // skip-GEMM out
    float* oout = (i == 0) ? R5 : (i == 1) ? R4 : (float*)d_out;  // stage out

    // 1) Patch-expand GEMM: [Mpre, Dp] x [Dp, 2Dp]
    {
      dim3 g((2 * Dp + BN - 1) / BN, Mpre / BM);
      gemm_bf16_wmma<<<g, 256, 0, stream>>>(bufPE, xcur, pe[i].w_exp, nullptr,
                                            Mpre, Dp, 2 * Dp);
    }
    // 2) Pixel-shuffle + LN + concat enc -> cat [T, 2D]
    shuffle_ln_cat<<<T / 8, 256, 0, stream>>>(bufCat, bufPE, enc[i],
                                              pe[i].ln_g, pe[i].ln_b, Hpre, D);
    // 3) Skip GEMM: [T, 2D] x [2D, D]
    {
      dim3 g((D + BN - 1) / BN, T / BM);
      gemm_bf16_wmma<<<g, 256, 0, stream>>>(xs, bufCat, sk[i].w, sk[i].b,
                                            T, 2 * D, D);
    }
    // 4) Q/K/V projections: [T, D] x [D, D]
    {
      dim3 g((D + BN - 1) / BN, T / BM);
      gemm_bf16_wmma<<<g, 256, 0, stream>>>(bufQ, xs, na[i].wq, na[i].bq, T, D, D);
      gemm_bf16_wmma<<<g, 256, 0, stream>>>(bufK, xs, na[i].wk, na[i].bk, T, D, D);
      gemm_bf16_wmma<<<g, 256, 0, stream>>>(bufV, xs, na[i].wv, na[i].bv, T, D, D);
    }
    // 5) Neighborhood attention core -> bufPE (reused)
    na_attend<<<T / 8, 256, 0, stream>>>(bufPE, bufQ, bufK, bufV, nbr[i],
                                         na[i].bias, Ntok, D);
    // 6) Output projection: [T, D] x [D, D]
    {
      dim3 g((D + BN - 1) / BN, T / BM);
      gemm_bf16_wmma<<<g, 256, 0, stream>>>(oout, bufPE, na[i].wo, na[i].bo,
                                            T, D, D);
    }
    xcur = oout;
  }
}
